// SelfAttention_180388626867
// MI455X (gfx1250) — matrix operand order repaired and checked by reading, not compiled, not run on hardware
//
#include <hip/hip_runtime.h>
#include <hip/hip_bf16.h>

// ---------------- problem constants ----------------
#define BATCH 8
#define CCH   256
#define NPIX  4096      // 64*64
#define DQK   32        // C/8
#define BQ    64        // query tile per block (4 waves x 16 rows)
#define BK    64        // key tile per iteration
#define NKB   (NPIX / BK)

typedef __attribute__((ext_vector_type(16))) __bf16 bf16x16;
typedef __attribute__((ext_vector_type(8)))  float  f32x8;
typedef __attribute__((ext_vector_type(4)))  int    v4i;

union Frag16 { bf16x16 v; unsigned u[8]; };

#if __has_builtin(__builtin_amdgcn_global_load_async_to_lds_b128) && \
    __has_builtin(__builtin_amdgcn_s_wait_asynccnt)
#define USE_ASYNC 1
#else
#define USE_ASYNC 0
#endif

// async-copy pointer casts: global (AS1) / LDS (AS3) v4i pointers
#define GPTR(p) ((__attribute__((address_space(1))) v4i*)(p))
#define LPTR(p) ((__attribute__((address_space(3))) v4i*)(p))

__device__ __forceinline__ unsigned short f2bf(float f) {
    unsigned u = __builtin_bit_cast(unsigned, f);
    unsigned r = (u + 0x7FFFu + ((u >> 16) & 1u)) >> 16;
    return (unsigned short)r;
}

// A matrix (16x32 bf16): rows striped across lanes, K pairs in VGPRs.
// lanes 0-15: V0..3 -> K 0..7, V4..7 -> K 16..23 ; lanes 16-31: +8 / +24.
__device__ __forceinline__ void load_a(Frag16& f, const unsigned short* base,
                                       int row_stride, int lane) {
    const int m = lane & 15, h = lane >> 4;
#pragma unroll
    for (int r = 0; r < 8; ++r) {
        const int k = ((r >> 2) << 4) + (h << 3) + ((r & 3) << 1);
        f.u[r] = *(const unsigned*)(base + m * row_stride + k);
    }
}

// B matrix (32x16 bf16): element B[k][n] = base[n*n_stride + k] (k pairs
// contiguous). lanes 0-15: V0..3 -> K 0..7, V4..7 -> K 16..23 ; lanes 16-31: +8 / +24.
__device__ __forceinline__ void load_b(Frag16& f, const unsigned short* base,
                                       int n_stride, int lane) {
    const int n = lane & 15, h = lane >> 4;
#pragma unroll
    for (int r = 0; r < 8; ++r) {
        const int k = ((r >> 2) << 4) + (h << 3) + ((r & 3) << 1);
        f.u[r] = *(const unsigned*)(base + n * n_stride + k);
    }
}

__device__ __forceinline__ f32x8 wmma_bf16(const Frag16& a, const Frag16& b,
                                           f32x8 c) {
    return __builtin_amdgcn_wmma_f32_16x16x32_bf16(
        false, a.v, false, b.v, (short)0, c, false, false);
}

// ---------------- kernel 0: weights f32 -> bf16 ----------------
__global__ void convert_weights(const float* __restrict__ Wq,
                                const float* __restrict__ Wk,
                                const float* __restrict__ Wv,
                                unsigned short* __restrict__ Wq_bf,
                                unsigned short* __restrict__ Wk_bf,
                                unsigned short* __restrict__ Wv_bf) {
    const int i = blockIdx.x * blockDim.x + threadIdx.x;
    const int NQ = DQK * CCH;            // 8192
    const int NV = CCH * CCH;            // 65536
    if (i < NQ) {
        Wq_bf[i] = f2bf(Wq[i]);
        Wk_bf[i] = f2bf(Wk[i]);
    }
    if (i < NV) Wv_bf[i] = f2bf(Wv[i]);
}

// ---------------- kernel 1: fused QKV projection (WMMA) ----------------
// X^T tile [64 pix][256 ch] staged bf16 in LDS; Q/K = [B,N,32], V = [B,C,N]
// (V transposed so the attention kernel can tile it with contiguous rows).
__launch_bounds__(128, 1)
__global__ void proj_kernel(const float* __restrict__ x,
                            const unsigned short* __restrict__ Wq_bf,
                            const unsigned short* __restrict__ Wk_bf,
                            const unsigned short* __restrict__ Wv_bf,
                            const float* __restrict__ bq,
                            const float* __restrict__ bk,
                            const float* __restrict__ bv,
                            unsigned short* __restrict__ Qb,
                            unsigned short* __restrict__ Kb,
                            unsigned short* __restrict__ Vt) {
    __shared__ unsigned short xs[BQ * CCH];   // [n_local][c], 32 KB

    const int tid  = threadIdx.x;
    const int wave = tid >> 5;
    const int lane = tid & 31;
    const int b    = blockIdx.y;
    const int n0   = blockIdx.x * BQ;

    for (int i = tid; i < BQ * CCH; i += 128) {
        const int n = i & (BQ - 1);
        const int c = i >> 6;
        const float xv = x[((size_t)(b * CCH + c)) * NPIX + n0 + n];
        xs[n * CCH + c] = f2bf(xv);
    }
    __syncthreads();

    f32x8 qacc[2] = {}, kacc[2] = {}, vacc[16] = {};

    const unsigned short* a_base = xs + (wave * 16) * CCH;
#pragma unroll 2
    for (int cc = 0; cc < CCH; cc += 32) {
        Frag16 a;
        load_a(a, a_base + cc, CCH, lane);
        Frag16 bw;
#pragma unroll
        for (int dg = 0; dg < 2; ++dg) {
            load_b(bw, Wq_bf + (dg * 16) * CCH + cc, CCH, lane);
            qacc[dg] = wmma_bf16(a, bw, qacc[dg]);
            load_b(bw, Wk_bf + (dg * 16) * CCH + cc, CCH, lane);
            kacc[dg] = wmma_bf16(a, bw, kacc[dg]);
        }
#pragma unroll
        for (int g = 0; g < 16; ++g) {
            load_b(bw, Wv_bf + (g * 16) * CCH + cc, CCH, lane);
            vacc[g] = wmma_bf16(a, bw, vacc[g]);
        }
    }

    // epilogue: bias add, bf16, scatter. C/D layout: row = r + 8*h, col = lane&15
    const int h  = lane >> 4;
    const int cl = lane & 15;
#pragma unroll
    for (int dg = 0; dg < 2; ++dg) {
        const int d = dg * 16 + cl;
        const float biq = bq[d], bik = bk[d];
#pragma unroll
        for (int r = 0; r < 8; ++r) {
            const int n = n0 + wave * 16 + r + 8 * h;
            Qb[((size_t)(b * NPIX + n)) * DQK + d] = f2bf(qacc[dg][r] + biq);
            Kb[((size_t)(b * NPIX + n)) * DQK + d] = f2bf(kacc[dg][r] + bik);
        }
    }
#pragma unroll
    for (int g = 0; g < 16; ++g) {
        const int o = g * 16 + cl;
        const float bio = bv[o];
#pragma unroll
        for (int r = 0; r < 8; ++r) {
            const int n = n0 + wave * 16 + r + 8 * h;
            Vt[((size_t)(b * CCH + o)) * NPIX + n] = f2bf(vacc[g][r] + bio);
        }
    }
}

// ---------------- kernel 2: fused flash attention (WMMA + async LDS) -------
__launch_bounds__(128, 1)
__global__ void attn_kernel(const unsigned short* __restrict__ Qb,
                            const unsigned short* __restrict__ Kb,
                            const unsigned short* __restrict__ Vt,
                            const float* __restrict__ x,
                            const float* __restrict__ gamma,
                            float* __restrict__ out) {
    __shared__ union SM {
        struct {
            unsigned short ks[2][BK * DQK];   // [key][d]   2 x 4 KB
            unsigned short vs[2][CCH * BK];   // [vc][key]  2 x 32 KB
            unsigned short ps[4][16 * BK];    // per-wave P 8 KB
        } s;
        float os[CCH][BQ + 1];                // epilogue O staging (65 KB)
    } sm;

    const int tid  = threadIdx.x;
    const int wave = tid >> 5;
    const int lane = tid & 31;
    const int b    = blockIdx.y;
    const int q0   = blockIdx.x * BQ;
    const int h    = lane >> 4;
    const int cl   = lane & 15;

    // Q A-fragment for this wave's 16 query rows (K dim = full D=32)
    Frag16 qa;
    load_a(qa, Qb + ((size_t)(b * NPIX + q0 + wave * 16)) * DQK, DQK, lane);

    f32x8 oacc[16] = {};
    float m_i[8], l_i[8];
#pragma unroll
    for (int r = 0; r < 8; ++r) { m_i[r] = -3.0e38f; l_i[r] = 0.0f; }

#if USE_ASYNC
    // async DMA of one K/V tile into LDS buffer `buf` (18 b128 ops / thread)
    auto issue_tile = [&](int kb, int buf) {
        const int k0 = kb * BK;
        const unsigned short* ksrc = Kb + ((size_t)(b * NPIX + k0)) * DQK;
#pragma unroll
        for (int it = 0; it < 2; ++it) {                       // K: 4 KB
            const int i = tid + it * 128;
            __builtin_amdgcn_global_load_async_to_lds_b128(
                GPTR(ksrc + i * 8),
                LPTR(sm.s.ks[buf] + i * 8), 0, 0);
        }
#pragma unroll
        for (int it = 0; it < 16; ++it) {                      // V: 32 KB
            const int i  = tid + it * 128;
            const int vc = i >> 3;
            const int o8 = (i & 7) * 8;
            __builtin_amdgcn_global_load_async_to_lds_b128(
                GPTR(Vt + ((size_t)(b * CCH + vc)) * NPIX + k0 + o8),
                LPTR(sm.s.vs[buf] + vc * BK + o8), 0, 0);
        }
    };
    issue_tile(0, 0);
#endif

    for (int kb = 0; kb < NKB; ++kb) {
#if USE_ASYNC
        const int cur = kb & 1;
        if (kb + 1 < NKB) {
            issue_tile(kb + 1, cur ^ 1);          // prefetch next tile
            __builtin_amdgcn_s_wait_asynccnt(18); // retire current tile only
        } else {
            __builtin_amdgcn_s_wait_asynccnt(0);
        }
#else
        const int cur = 0;
        {
            const int k0 = kb * BK;
            const unsigned* ks32 =
                (const unsigned*)(Kb + ((size_t)(b * NPIX + k0)) * DQK);
            unsigned* kd = (unsigned*)sm.s.ks[0];
            for (int i = tid; i < (BK * DQK) / 2; i += 128) kd[i] = ks32[i];
            unsigned* vd = (unsigned*)sm.s.vs[0];
            for (int i = tid; i < (BK * CCH) / 2; i += 128) {
                const int vc = i >> 5, o = i & 31;
                vd[vc * 32 + o] =
                    *(const unsigned*)(Vt + ((size_t)(b * CCH + vc)) * NPIX +
                                       k0 + 2 * o);
            }
            if (kb + 1 < NKB) {
                const int nk0 = k0 + BK;
                __builtin_prefetch(Vt + (size_t)b * CCH * NPIX + nk0 +
                                       (size_t)tid * 2 * NPIX, 0, 0);
                __builtin_prefetch(Kb + ((size_t)(b * NPIX + nk0)) * DQK +
                                       (size_t)tid * 16, 0, 0);
            }
        }
#endif
        __syncthreads();

        const unsigned short* ksb = sm.s.ks[cur];
        const unsigned short* vsb = sm.s.vs[cur];

        // ---- S = Q K^T : 16 x 64, four 16x16 WMMAs ----
        f32x8 s[4];
#pragma unroll
        for (int j = 0; j < 4; ++j) {
            Frag16 bk_;
            load_b(bk_, ksb + (j * 16) * DQK, DQK, lane);
            f32x8 z = {};
            s[j] = wmma_bf16(qa, bk_, z);
        }

        // ---- online softmax (each row lives in one 16-lane half) ----
        float mb[8];
#pragma unroll
        for (int r = 0; r < 8; ++r) {
            float m = s[0][r];
#pragma unroll
            for (int j = 1; j < 4; ++j) m = fmaxf(m, s[j][r]);
            m = fmaxf(m, __shfl_xor(m, 8, 32));
            m = fmaxf(m, __shfl_xor(m, 4, 32));
            m = fmaxf(m, __shfl_xor(m, 2, 32));
            m = fmaxf(m, __shfl_xor(m, 1, 32));
            mb[r] = m;
        }

        float scale[8];
#pragma unroll
        for (int r = 0; r < 8; ++r) {
            const float mn = fmaxf(m_i[r], mb[r]);
            scale[r] = __expf(m_i[r] - mn);
            m_i[r] = mn;
        }

        float lb[8];
#pragma unroll
        for (int r = 0; r < 8; ++r) lb[r] = 0.0f;
#pragma unroll
        for (int j = 0; j < 4; ++j) {
#pragma unroll
            for (int r = 0; r < 8; ++r) {
                const float p = __expf(s[j][r] - m_i[r]);
                lb[r] += p;
                sm.s.ps[wave][(r + 8 * h) * BK + j * 16 + cl] = f2bf(p);
            }
        }
#pragma unroll
        for (int r = 0; r < 8; ++r) {
            float l = lb[r];
            l += __shfl_xor(l, 8, 32);
            l += __shfl_xor(l, 4, 32);
            l += __shfl_xor(l, 2, 32);
            l += __shfl_xor(l, 1, 32);
            l_i[r] = l_i[r] * scale[r] + l;
        }

#pragma unroll
        for (int g = 0; g < 16; ++g)
#pragma unroll
            for (int r = 0; r < 8; ++r) oacc[g][r] *= scale[r];

        // ---- O += P V : 32 WMMAs ----
        Frag16 pa0, pa1;
        load_a(pa0, sm.s.ps[wave], BK, lane);
        load_a(pa1, sm.s.ps[wave] + 32, BK, lane);
#pragma unroll
        for (int g = 0; g < 16; ++g) {
            Frag16 bv0, bv1;
            load_b(bv0, vsb + (g * 16) * BK, BK, lane);
            load_b(bv1, vsb + (g * 16) * BK + 32, BK, lane);
            oacc[g] = wmma_bf16(pa0, bv0, oacc[g]);
            oacc[g] = wmma_bf16(pa1, bv1, oacc[g]);
        }
        __syncthreads();
    }

    // ---- epilogue: stage O in LDS, then coalesced out = gamma*O/l + x ----
    float inv[8];
#pragma unroll
    for (int r = 0; r < 8; ++r) inv[r] = 1.0f / l_i[r];
#pragma unroll
    for (int g = 0; g < 16; ++g) {
        const int c = g * 16 + cl;
#pragma unroll
        for (int r = 0; r < 8; ++r)
            sm.os[c][wave * 16 + r + 8 * h] = oacc[g][r] * inv[r];
    }
    __syncthreads();

    const float g0 = gamma[0];
    for (int i = tid; i < CCH * BQ; i += 128) {
        const int n = i & (BQ - 1);
        const int c = i >> 6;
        const size_t idx = ((size_t)(b * CCH + c)) * NPIX + q0 + n;
        out[idx] = g0 * sm.os[c][n] + x[idx];
    }
}

// ---------------- host launcher ----------------
extern "C" void kernel_launch(void* const* d_in, const int* in_sizes, int n_in,
                              void* d_out, int out_size, void* d_ws, size_t ws_size,
                              hipStream_t stream) {
    (void)in_sizes; (void)n_in; (void)out_size; (void)ws_size;
    const float* x     = (const float*)d_in[0];
    const float* Wq    = (const float*)d_in[1];
    const float* bq    = (const float*)d_in[2];
    const float* Wk    = (const float*)d_in[3];
    const float* bk    = (const float*)d_in[4];
    const float* Wv    = (const float*)d_in[5];
    const float* bv    = (const float*)d_in[6];
    const float* gamma = (const float*)d_in[7];
    float* out = (float*)d_out;

    // workspace layout (bf16 as u16)
    unsigned short* Qb    = (unsigned short*)d_ws;
    unsigned short* Kb    = Qb + (size_t)BATCH * NPIX * DQK;
    unsigned short* Vt    = Kb + (size_t)BATCH * NPIX * DQK;   // [B][C][N]
    unsigned short* Wq_bf = Vt + (size_t)BATCH * NPIX * CCH;
    unsigned short* Wk_bf = Wq_bf + DQK * CCH;
    unsigned short* Wv_bf = Wk_bf + DQK * CCH;

    convert_weights<<<(CCH * CCH + 255) / 256, 256, 0, stream>>>(
        Wq, Wk, Wv, Wq_bf, Wk_bf, Wv_bf);

    proj_kernel<<<dim3(NPIX / BQ, BATCH), 128, 0, stream>>>(
        x, Wq_bf, Wk_bf, Wv_bf, bq, bk, bv, Qb, Kb, Vt);

    attn_kernel<<<dim3(NPIX / BQ, BATCH), 128, 0, stream>>>(
        Qb, Kb, Vt, x, gamma, out);
}
